// MessagePassingLayer_17995912970761
// MI455X (gfx1250) — matrix operand order repaired and checked by reading, not compiled, not run on hardware
//
#include <hip/hip_runtime.h>

#define Bb 16
#define Nn 256
#define Ff 64
#define Hh 128
#define HPAD 136   // 128 + 8 halves: 272B row stride -> 4-bank rotation per row, no LDS conflicts

typedef __attribute__((ext_vector_type(8)))  _Float16 v8h;
typedef __attribute__((ext_vector_type(16))) _Float16 v16h;
typedef __attribute__((ext_vector_type(8)))  float    v8f;

__device__ __forceinline__ float softplus_f(float x) {
    // numerically stable: max(x,0) + log(1 + exp(-|x|))
    float e = __expf(-fabsf(x));
    return fmaxf(x, 0.0f) + __logf(1.0f + e);
}

__device__ __forceinline__ v16h cat16(v8h lo, v8h hi) {
    return __builtin_shufflevector(lo, hi, 0,1,2,3,4,5,6,7,8,9,10,11,12,13,14,15);
}

__global__ __launch_bounds__(256)
void mp_layer_wmma_kernel(const float* __restrict__ r_batch,
                          const float* __restrict__ f_batch,
                          const float* __restrict__ w1,
                          const float* __restrict__ b1,
                          const float* __restrict__ W2,
                          const float* __restrict__ b2,
                          float* __restrict__ out) {
    __shared__ _Float16 sh_h[128][HPAD];     // one half (128 rows) of h[j][:] in f16
    __shared__ _Float16 sh_w2t[Ff][HPAD];    // W2 transposed [F][H] in f16
    __shared__ float    sh_d[Nn];
    __shared__ float    sh_w1[Hh];
    __shared__ float    sh_b1[Hh];
    __shared__ float    sh_b2[Ff];
    __shared__ float    sh_red[8][32];

    const int tid  = threadIdx.x;
    const int lane = tid & 31;
    const int wave = tid >> 5;
    const int bi   = blockIdx.x;
    const int b    = bi / Nn;
    const int i    = bi % Nn;

    // stage small parameter vectors
    if (tid < Hh) { sh_w1[tid] = w1[tid]; sh_b1[tid] = b1[tid]; }
    if (tid < Ff) { sh_b2[tid] = b2[tid]; }

    // W2 [H,F] -> sh_w2t [F][H] (f16)
    for (int idx = tid; idx < Hh * Ff; idx += 256) {
        int hh = idx >> 6;      // row of W2
        int ff = idx & 63;      // col of W2
        sh_w2t[ff][hh] = (_Float16)W2[idx];
    }

    // pairwise distances: row i vs all j (one j per thread)
    {
        const float* ri = r_batch + (size_t)(b * Nn + i) * 3;
        const float* rj = r_batch + (size_t)(b * Nn + tid) * 3;
        float dx = ri[0] - rj[0];
        float dy = ri[1] - rj[1];
        float dz = ri[2] - rj[2];
        float s  = dx * dx + dy * dy + dz * dz;
        sh_d[tid] = (s > 0.0f) ? sqrtf(s) : 0.0f;
    }
    __syncthreads();

    // wave -> (f-tile, j-group) assignment
    const int ft = wave & 3;            // f-tile: cols f0..f0+15
    const int jg = wave >> 2;           // j-group within a pass (0/1)
    const int f0 = ft * 16;
    const int fl = lane & 15;
    const int kh = lane >> 4;           // lane half

    // B fragments (16x16x32 f16 layout): element e<8 <-> K = k0 + kh*8 + e,
    // e>=8 <-> K = k0 + 16 + kh*8 + (e-8), N = lane&15. Loaded once, reused.
    v16h bfrag[4];
    #pragma unroll
    for (int kk = 0; kk < 4; ++kk) {
        const _Float16* p = &sh_w2t[f0 + fl][kk * 32 + kh * 8];
        bfrag[kk] = cat16(*(const v8h*)p, *(const v8h*)(p + 16));
    }
    const float b2f = sh_b2[f0 + fl];

    float rsum = 0.0f;   // running sum over j of softplus(d')*f  (fixed f per lane)

    for (int pass = 0; pass < 2; ++pass) {
        // compute h rows [pass*128 .. +127]: 2 threads per row, 64 cols each
        {
            int jr = tid & 127;
            int ch = tid >> 7;
            float dj = sh_d[pass * 128 + jr];
            #pragma unroll 8
            for (int c = 0; c < 64; ++c) {
                int col = ch * 64 + c;
                float x = fmaf(dj, sh_w1[col], sh_b1[col]);
                sh_h[jr][col] = (_Float16)softplus_f(x);
            }
        }
        __syncthreads();

        // wave processes 4 j-tiles of this pass
        #pragma unroll
        for (int t = 0; t < 4; ++t) {
            int j0l = (jg * 4 + t) * 16;
            v8f acc = {};
            #pragma unroll
            for (int kk = 0; kk < 4; ++kk) {
                // A fragment (16x32 f16): row = j0l + (lane&15),
                // cols: [k0 + kh*8 .. +7] and [k0 + kh*8 + 16 .. +23]
                const _Float16* p = &sh_h[j0l + fl][kk * 32 + kh * 8];
                v16h a = cat16(*(const v8h*)p, *(const v8h*)(p + 16));
                acc = __builtin_amdgcn_wmma_f32_16x16x32_f16(
                        false, a, false, bfrag[kk], (short)0, acc, false, false);
            }
            // epilogue: d' = softplus(acc + b2); rsum += d' * f_batch[b,j,f]
            // C/D layout: VGPR k <-> row m = k + 8*kh, col = f0 + fl
            int jbase = pass * 128 + j0l + kh * 8;
            const float* fp = f_batch + (size_t)(b * Nn + jbase) * Ff + f0 + fl;
            #pragma unroll
            for (int k = 0; k < 8; ++k) {
                float dp = softplus_f(acc[k] + b2f);
                rsum = fmaf(dp, fp[(size_t)k * Ff], rsum);
            }
        }
        __syncthreads();   // before sh_h is overwritten by next pass
    }

    // reduce across lane halves and the 2 waves sharing each f-tile
    sh_red[wave][lane] = rsum;
    __syncthreads();

    if (tid < Ff) {
        int wft = tid >> 4;
        int wfl = tid & 15;
        float tot = sh_red[wft][wfl]     + sh_red[wft][wfl + 16]
                  + sh_red[wft + 4][wfl] + sh_red[wft + 4][wfl + 16];
        out[(size_t)(b * Nn + i) * Ff + tid] = softplus_f(tot);
    }
}

extern "C" void kernel_launch(void* const* d_in, const int* in_sizes, int n_in,
                              void* d_out, int out_size, void* d_ws, size_t ws_size,
                              hipStream_t stream) {
    (void)in_sizes; (void)n_in; (void)out_size; (void)d_ws; (void)ws_size;
    const float* r_batch = (const float*)d_in[0];
    const float* f_batch = (const float*)d_in[1];
    const float* w1      = (const float*)d_in[2];
    const float* b1      = (const float*)d_in[3];
    const float* W2      = (const float*)d_in[4];
    const float* b2      = (const float*)d_in[5];
    float* out = (float*)d_out;

    dim3 grid(Bb * Nn);   // one workgroup per (b, i)
    dim3 block(256);      // 8 wave32
    mp_layer_wmma_kernel<<<grid, block, 0, stream>>>(r_batch, f_batch, w1, b1, W2, b2, out);
}
